// ActorCriticRange_60224031425148
// MI455X (gfx1250) — matrix operand order repaired and checked by reading, not compiled, not run on hardware
//
#include <hip/hip_runtime.h>
#include <hip/hip_bf16.h>

// ---------------------------------------------------------------------------
// MI455X (gfx1250) ActorCritic forward.
// All big matmuls: V_WMMA_F32_16X16X32_F16 on pre-split (hi,lo) f16 operands
// (Ahi*Bhi + Ahi*Blo + Alo*Bhi ~ fp32 precision, f32 accumulate).
// Operands pre-padded (rows%32==0, K%32==0, per-expert O padded) so the
// WMMA K-loop is branch-free: 16x b128 loads + 12 WMMAs per K=32 step.
// ---------------------------------------------------------------------------

typedef __attribute__((ext_vector_type(16))) _Float16 v16h;
typedef __attribute__((ext_vector_type(8)))  _Float16 v8h;
typedef __attribute__((ext_vector_type(8)))  float    v8f;

#define WMMA_F16(A, B, C) \
  __builtin_amdgcn_wmma_f32_16x16x32_f16(false, (A), false, (B), (short)0, (C), false, false)

// A-frag (16x32 f16): lane l -> row l&15, 16 contiguous halves at k=(l>>4)*16.
__device__ __forceinline__ v16h ldfragA(const _Float16* p) {
  union { v16h v; v8h h[2]; } u;
  u.h[0] = *(const v8h*)p;
  u.h[1] = *(const v8h*)(p + 8);
  return u.v;
}
// B-frag (32x16 f16): lane l -> W-row col0+(l&15), 16 contiguous halves at (l>>4)*16.
__device__ __forceinline__ v16h ldfragB(const _Float16* p) {
  union { v16h v; v8h h[2]; } u;
  u.h[0] = *(const v8h*)p;
  u.h[1] = *(const v8h*)(p + 8);
  return u.v;
}

// Y[m][r,o] = sum_k A[m][r,k]*W[m][o,k] + bias[m][o]
// One wave -> 32x32 output tile (2x2 WMMA tiles, 4 indep accumulators).
// A/W are pre-split f16 (hi,lo), rows padded to 32, Kp%32==0 -> no guards in loop.
__global__ __launch_bounds__(128) void gemm_h2_wmma(
    const _Float16* __restrict__ Ahi, const _Float16* __restrict__ Alo, long lda, long sXm,
    const _Float16* __restrict__ Whi, const _Float16* __restrict__ Wlo, long sWm,
    const float* __restrict__ bias, long sBm,
    float* __restrict__ Y, long ldy, long sYm,
    _Float16* __restrict__ Yhi, _Float16* __restrict__ Ylo, long ldyh, long sYhm,
    int R, int O, int Kp)
{
  int lane = threadIdx.x & 31, wave = threadIdx.x >> 5;
  int tn2 = blockIdx.y * 4 + wave;
  if (tn2 * 32 >= O) return;                 // wave-uniform: EXEC stays full
  int tm2 = blockIdx.x, m = blockIdx.z;
  int row0 = tm2 * 32, col0 = tn2 * 32;
  int lr = lane & 15, hl = lane >> 4;

  const _Float16* Ah = Ahi + (long)m * sXm;
  const _Float16* Al = Alo + (long)m * sXm;
  const _Float16* Bh = Whi + (long)m * sWm;
  const _Float16* Bl = Wlo + (long)m * sWm;
  const _Float16* a0h = Ah + (long)(row0 + lr) * lda + hl * 16;
  const _Float16* a0l = Al + (long)(row0 + lr) * lda + hl * 16;
  const _Float16* a1h = Ah + (long)(row0 + 16 + lr) * lda + hl * 16;
  const _Float16* a1l = Al + (long)(row0 + 16 + lr) * lda + hl * 16;
  const _Float16* b0h = Bh + (long)(col0 + lr) * Kp + hl * 16;
  const _Float16* b0l = Bl + (long)(col0 + lr) * Kp + hl * 16;
  const _Float16* b1h = Bh + (long)(col0 + 16 + lr) * Kp + hl * 16;
  const _Float16* b1l = Bl + (long)(col0 + 16 + lr) * Kp + hl * 16;

  v8f acc[2][2] = {};
  for (int k0 = 0; k0 < Kp; k0 += 32) {
    v16h A0h = ldfragA(a0h + k0), A0l = ldfragA(a0l + k0);
    v16h A1h = ldfragA(a1h + k0), A1l = ldfragA(a1l + k0);
    v16h B0h = ldfragB(b0h + k0), B0l = ldfragB(b0l + k0);
    v16h B1h = ldfragB(b1h + k0), B1l = ldfragB(b1l + k0);
    acc[0][0] = WMMA_F16(A0h, B0h, acc[0][0]);
    acc[0][1] = WMMA_F16(A0h, B1h, acc[0][1]);
    acc[1][0] = WMMA_F16(A1h, B0h, acc[1][0]);
    acc[1][1] = WMMA_F16(A1h, B1h, acc[1][1]);
    acc[0][0] = WMMA_F16(A0h, B0l, acc[0][0]);
    acc[0][1] = WMMA_F16(A0h, B1l, acc[0][1]);
    acc[1][0] = WMMA_F16(A1h, B0l, acc[1][0]);
    acc[1][1] = WMMA_F16(A1h, B1l, acc[1][1]);
    acc[0][0] = WMMA_F16(A0l, B0h, acc[0][0]);
    acc[0][1] = WMMA_F16(A0l, B1h, acc[0][1]);
    acc[1][0] = WMMA_F16(A1l, B0h, acc[1][0]);
    acc[1][1] = WMMA_F16(A1l, B1h, acc[1][1]);
  }
  // Epilogue: C layout lane l -> col = (l&15), VGPR r -> row r + 8*(l>>4).
#pragma unroll
  for (int ni = 0; ni < 2; ++ni) {
    int col = col0 + ni * 16 + lr;
    if (col >= O) continue;
    float bv = bias ? bias[(long)m * sBm + col] : 0.f;
#pragma unroll
    for (int mi = 0; mi < 2; ++mi) {
      int rbase = row0 + mi * 16 + hl * 8;
#pragma unroll
      for (int r = 0; r < 8; ++r) {
        int row = rbase + r;
        if (row >= R) continue;
        float v = acc[mi][ni][r] + bv;
        if (Y)   Y[(long)m * sYm + (long)row * ldy + col] = v;
        if (Yhi) {
          long idx = (long)m * sYhm + (long)row * ldyh + col;
          _Float16 h = (_Float16)v;
          Yhi[idx] = h; Ylo[idx] = (_Float16)(v - (float)h);
        }
      }
    }
  }
}

// Generic f32 -> (hi,lo) f16 with group-wise row/col zero-padding.
__global__ void split_pad_kernel(const float* __restrict__ src, int rin, int cin, long ldin,
                                 int ngrp, int rout, int cout,
                                 _Float16* __restrict__ hi, _Float16* __restrict__ lo)
{
  long id = (long)blockIdx.x * blockDim.x + threadIdx.x;
  long per = (long)rout * cout;
  if (id >= (long)ngrp * per) return;
  int g = (int)(id / per);
  long rem = id - (long)g * per;
  int r = (int)(rem / cout), c = (int)(rem % cout);
  float v = (r < rin && c < cin) ? src[((long)g * rin + r) * ldin + c] : 0.f;
  _Float16 h = (_Float16)v;
  hi[id] = h; lo[id] = (_Float16)(v - (float)h);
}

__device__ __forceinline__ float gelu_tanh(float v) {
  float v3 = v * v * v;
  return 0.5f * v * (1.f + tanhf(0.7978845608028654f * (v + 0.044715f * v3)));
}

// Rowwise LayerNorm (+optional tanh-GELU); writes f32 in-place and/or (hi,lo) f16.
__global__ __launch_bounds__(128) void ln_act_kernel(
    float* __restrict__ X, const float* __restrict__ gg, const float* __restrict__ bb,
    int D, int R, int act, int wf32,
    _Float16* __restrict__ hi, _Float16* __restrict__ lo)
{
  int row = blockIdx.x;
  if (row >= R) return;
  float* x = X + (long)row * D;
  int tid = threadIdx.x, lane = tid & 31, wv = tid >> 5;
  float s = 0.f, ss = 0.f;
  for (int i = tid; i < D; i += 128) { float v = x[i]; s += v; ss += v * v; }
#pragma unroll
  for (int o = 16; o; o >>= 1) { s += __shfl_xor(s, o); ss += __shfl_xor(ss, o); }
  __shared__ float rs[4], rss[4];
  if (lane == 0) { rs[wv] = s; rss[wv] = ss; }
  __syncthreads();
  s  = rs[0] + rs[1] + rs[2] + rs[3];
  ss = rss[0] + rss[1] + rss[2] + rss[3];
  float mu  = s / D;
  float var = fmaxf(ss / D - mu * mu, 0.f);
  float inv = rsqrtf(var + 1e-5f);
  for (int i = tid; i < D; i += 128) {
    float v = (x[i] - mu) * inv * gg[i] + bb[i];
    if (act) v = gelu_tanh(v);
    if (wf32) x[i] = v;
    if (hi) {
      long idx = (long)row * D + i;
      _Float16 h = (_Float16)v;
      hi[idx] = h; lo[idx] = (_Float16)(v - (float)h);
    }
  }
}

// qk[h][c] = sum_d qh[h*128+d]*Wk[h*128+d][c]; qk[h][512] = qh_h . bk_h
__global__ void qk_kernel(const float* __restrict__ qh, const float* __restrict__ Wk,
                          const float* __restrict__ bk, float* __restrict__ qk)
{
  int id = blockIdx.x * blockDim.x + threadIdx.x;
  if (id >= 4 * 513) return;
  int h = id / 513, c = id % 513;
  const float* q = qh + h * 128;
  float s = 0.f;
  if (c < 512) {
    for (int d = 0; d < 128; ++d) s += q[d] * Wk[(long)(h * 128 + d) * 512 + c];
    qk[h * 520 + c] = s;
  } else {
    for (int d = 0; d < 128; ++d) s += q[d] * bk[h * 128 + d];
    qk[h * 520 + 512] = s;
  }
}

// s[b,h,n] = (kv2[bl,n,:] . qk[h,:] + qkb[h]) / sqrt(128)
__global__ void scores_kernel(const float* __restrict__ kv2, const float* __restrict__ qk,
                              float* __restrict__ s, int b0, int nb)
{
  int id = blockIdx.x * blockDim.x + threadIdx.x;
  if (id >= nb * 256) return;
  int bl = id >> 8, h = (id >> 6) & 3, n = id & 63;
  const float* kvp = kv2 + ((long)bl * 64 + n) * 512;
  const float* qp  = qk + h * 520;
  float acc = qp[512];
  for (int c = 0; c < 512; ++c) acc += kvp[c] * qp[c];
  s[(((long)(b0 + bl)) * 4 + h) * 64 + n] = acc * 0.08838834764831845f;
}

// softmax over n=64 per (b,h); one wave per row.
__global__ __launch_bounds__(128) void softmax64_kernel(float* __restrict__ s, int b0, int nb)
{
  int lane = threadIdx.x & 31, wv = threadIdx.x >> 5;
  int row = blockIdx.x * 4 + wv;
  if (row >= nb * 4) return;
  float* p = s + ((long)b0 * 4 + row) * 64;
  float v0 = p[lane], v1 = p[lane + 32];
  float mx = fmaxf(v0, v1);
#pragma unroll
  for (int o = 16; o; o >>= 1) mx = fmaxf(mx, __shfl_xor(mx, o));
  float e0 = expf(v0 - mx), e1 = expf(v1 - mx);
  float sm = e0 + e1;
#pragma unroll
  for (int o = 16; o; o >>= 1) sm += __shfl_xor(sm, o);
  float r = 1.f / sm;
  p[lane] = e0 * r; p[lane + 32] = e1 * r;
}

// w[b,h,c] = sum_n att[b,h,n]*kv2[bl,n,c]  -> written directly as (hi,lo) f16
__global__ __launch_bounds__(128) void context_kernel(
    const float* __restrict__ kv2, const float* __restrict__ s,
    _Float16* __restrict__ whi, _Float16* __restrict__ wlo, int b0)
{
  int bl = blockIdx.x, h = blockIdx.y, tid = threadIdx.x;
  __shared__ float att[64];
  if (tid < 64) att[tid] = s[(((long)(b0 + bl)) * 4 + h) * 64 + tid];
  __syncthreads();
#pragma unroll
  for (int j = 0; j < 4; ++j) {
    int c = tid + j * 128;
    float acc = 0.f;
    for (int n = 0; n < 64; ++n) acc += att[n] * kv2[((long)bl * 64 + n) * 512 + c];
    long idx = ((long)(b0 + bl)) * 2048 + h * 512 + c;
    _Float16 hh = (_Float16)acc;
    whi[idx] = hh; wlo[idx] = (_Float16)(acc - (float)hh);
  }
}

// ctx = [state(48) | emb(128) | zero-pad to 192] -> (hi,lo) f16
__global__ void concat_split_kernel(const float* __restrict__ st, const float* __restrict__ emb,
                                    _Float16* __restrict__ hi, _Float16* __restrict__ lo)
{
  long id = (long)blockIdx.x * blockDim.x + threadIdx.x;
  if (id >= 2048L * 192) return;
  int b = (int)(id / 192), i = (int)(id % 192);
  float v = (i < 48) ? st[b * 48 + i] : ((i < 176) ? emb[b * 128 + (i - 48)] : 0.f);
  _Float16 h = (_Float16)v;
  hi[id] = h; lo[id] = (_Float16)(v - (float)h);
}

__global__ void softmax8_kernel(float* __restrict__ pg)
{
  int id = blockIdx.x * blockDim.x + threadIdx.x;
  if (id >= 2048 * 8) return;
  float* p = pg + (long)id * 8;
  float mx = p[0];
#pragma unroll
  for (int j = 1; j < 8; ++j) mx = fmaxf(mx, p[j]);
  float e[8], sm = 0.f;
#pragma unroll
  for (int j = 0; j < 8; ++j) { e[j] = expf(p[j] - mx); sm += e[j]; }
  float r = 1.f / sm;
#pragma unroll
  for (int j = 0; j < 8; ++j) p[j] = e[j] * r;
}

// x_next = elu((sum_m gw*h_m)*s1 + (sum_m gb*ab_m)*s2); l<3 -> (hi,lo) f16, l==3 -> f32 out
__global__ void moe_combine_kernel(const float* __restrict__ h, const float* __restrict__ pg,
                                   const float* __restrict__ ab, const float* __restrict__ sc,
                                   _Float16* __restrict__ xhi, _Float16* __restrict__ xlo,
                                   float* __restrict__ xf, int O, int l)
{
  long id = (long)blockIdx.x * blockDim.x + threadIdx.x;
  if (id >= 2048L * O) return;
  int b = (int)(id / O), o = (int)(id % O);
  const float* g = pg + (long)b * 64 + (2 * l) * 8;
  float y = 0.f, yb = 0.f;
#pragma unroll
  for (int m = 0; m < 8; ++m) {
    y  += g[m]     * h[(long)m * 2048 * O + id];
    yb += g[8 + m] * ab[m * O + o];
  }
  if (l < 3) {
    float s1 = 1.f + sc[(long)b * 1536 + (2 * l)     * 256 + o];
    float s2 = 1.f + sc[(long)b * 1536 + (2 * l + 1) * 256 + o];
    float v = y * s1 + yb * s2;
    v = (v > 0.f) ? v : (expf(v) - 1.f);
    _Float16 hh = (_Float16)v;
    xhi[id] = hh; xlo[id] = (_Float16)(v - (float)hh);
  } else {
    xf[id] = y + yb;
  }
}

// ---------------------------------------------------------------------------
// Host side
// ---------------------------------------------------------------------------

struct HB { _Float16 *hi, *lo; };
static const HB HBNULL = { nullptr, nullptr };

static inline void gemm(HB A, long lda, long sXm, HB W, long sWm,
                        const float* bias, long sBm,
                        float* Y, long ldy, long sYm,
                        HB Yh, long ldyh, long sYhm,
                        int R, int O, int Kp, int nM, hipStream_t st)
{
  dim3 grid((R + 31) / 32, (((O + 31) / 32) + 3) / 4, nM);
  gemm_h2_wmma<<<grid, 128, 0, st>>>(A.hi, A.lo, lda, sXm, W.hi, W.lo, sWm,
                                     bias, sBm, Y, ldy, sYm,
                                     Yh.hi, Yh.lo, ldyh, sYhm, R, O, Kp);
}

static inline void split(const float* src, int rin, int cin, int ngrp, int rout, int cout,
                         HB dst, hipStream_t st)
{
  long tot = (long)ngrp * rout * cout;
  split_pad_kernel<<<(unsigned)((tot + 255) / 256), 256, 0, st>>>(
      src, rin, cin, cin, ngrp, rout, cout, dst.hi, dst.lo);
}

struct FuseP {
  HB wkv1, wkv2, wq, wo, wv, qt, wpl1, wpl2;
  const float *l1b, *l2b, *ln1b, *ln1g, *ln2b, *ln2g;
  const float *kW, *kb, *ob, *qb, *vb;
  const float *pl1b, *pl2b, *pln1b, *pln1g;
};
struct FuseWs {
  float *s, *qh, *qk, *p1, *kv1, *kv2;
  HB embc, kv1h, wctx, atto, oprj, p1h;
};

static void run_fuse(const float* obs_embed, const FuseP& P, const FuseWs& W,
                     float* emb, hipStream_t st)
{
  // qh = qtok @ Wq^T + bq (batch-invariant); qk folds Wk into the query side.
  gemm(P.qt, 512, 0, P.wq, 0, P.qb, 0, W.qh, 512, 0, HBNULL, 0, 0, 1, 512, 512, 1, st);
  qk_kernel<<<9, 256, 0, st>>>(W.qh, P.kW, P.kb, W.qk);

  // kv pipeline chunked (256 batches * 64 tokens) so panels stay L2-resident.
  for (int c = 0; c < 8; ++c) {
    int b0 = c * 256;
    split(obs_embed + (long)b0 * 64 * 128, 16384, 128, 1, 16384, 128, W.embc, st);
    gemm(W.embc, 128, 0, P.wkv1, 0, P.l1b, 0, W.kv1, 320, 0, HBNULL, 0, 0,
         16384, 320, 128, 1, st);
    ln_act_kernel<<<16384, 128, 0, st>>>(W.kv1, P.ln1g, P.ln1b, 320, 16384, 1, 0,
                                         W.kv1h.hi, W.kv1h.lo);
    gemm(W.kv1h, 320, 0, P.wkv2, 0, P.l2b, 0, W.kv2, 512, 0, HBNULL, 0, 0,
         16384, 512, 320, 1, st);
    ln_act_kernel<<<16384, 128, 0, st>>>(W.kv2, P.ln2g, P.ln2b, 512, 16384, 1, 1,
                                         nullptr, nullptr);
    scores_kernel<<<256, 256, 0, st>>>(W.kv2, W.qk, W.s, b0, 256);
    softmax64_kernel<<<256, 128, 0, st>>>(W.s, b0, 256);
    context_kernel<<<dim3(256, 4), 128, 0, st>>>(W.kv2, W.s, W.wctx.hi, W.wctx.lo, b0);
  }

  // attO[b,h*128+d] = w[b,h,:] @ Wv_h^T + bv_h  (batched over 4 heads, f16 epilogue)
  gemm(W.wctx, 2048, 512, P.wv, 128L * 512, P.vb, 128, nullptr, 0, 0,
       W.atto, 512, 128, 2048, 128, 512, 4, st);
  gemm(W.atto, 512, 0, P.wo, 0, P.ob, 0, nullptr, 0, 0, W.oprj, 512, 0,
       2048, 512, 512, 1, st);
  gemm(W.oprj, 512, 0, P.wpl1, 0, P.pl1b, 0, W.p1, 320, 0, HBNULL, 0, 0,
       2048, 320, 512, 1, st);
  ln_act_kernel<<<2048, 128, 0, st>>>(W.p1, P.pln1g, P.pln1b, 320, 2048, 1, 0,
                                      W.p1h.hi, W.p1h.lo);
  gemm(W.p1h, 320, 0, P.wpl2, 0, P.pl2b, 0, emb, 128, 0, HBNULL, 0, 0,
       2048, 128, 320, 1, st);
}

extern "C" void kernel_launch(void* const* d_in, const int* in_sizes, int n_in,
                              void* d_out, int out_size, void* d_ws, size_t ws_size,
                              hipStream_t stream)
{
  (void)in_sizes; (void)n_in; (void)out_size; (void)ws_size;
#define IN(i) ((const float*)d_in[(i)])
  // jax tree-leaves order (dict keys sorted at every level):
  const float* obs_embed = IN(0);
  const float* obs_state = IN(1);
  const float* aW[4] = { IN(2), IN(3), IN(4), IN(5) };
  const float* ab[4] = { IN(6), IN(7), IN(8), IN(9) };
  const float *bb1W = IN(10), *bb1b = IN(11), *bb2W = IN(12), *bb2b = IN(13);
  const float *bbln1b = IN(14), *bbln1g = IN(15), *bbln2b = IN(16), *bbln2g = IN(17);
  const float *logW = IN(18), *logb = IN(19), *scW = IN(20), *scb = IN(21);

  FuseP M, S;
  M.l1b = IN(23); M.l2b = IN(25); M.ln1b = IN(26); M.ln1g = IN(27);
  M.ln2b = IN(28); M.ln2g = IN(29);
  S.l1b = IN(31); S.l2b = IN(33); S.ln1b = IN(34); S.ln1g = IN(35);
  S.ln2b = IN(36); S.ln2g = IN(37);
  M.kW = IN(38); M.kb = IN(39); M.ob = IN(41); M.qb = IN(43); M.vb = IN(45);
  S.kW = IN(46); S.kb = IN(47); S.ob = IN(49); S.qb = IN(51); S.vb = IN(53);
  M.pl1b = S.pl1b = IN(55); M.pl2b = S.pl2b = IN(57);
  M.pln1b = S.pln1b = IN(58); M.pln1g = S.pln1g = IN(59);

  // ---- workspace carve-out ----
  float* wsf = (float*)d_ws;
  long fo = 0;
  auto allocF = [&](long n) { float* p = wsf + fo; fo += (n + 15) & ~15L; return p; };
  float* s_att = allocF(2048L * 4 * 64);
  float* qh    = allocF(512);
  float* qk    = allocF(4 * 520);
  float* p1    = allocF(2048L * 320);
  float* emb   = allocF(2048L * 128);
  float* z1    = allocF(2048L * 512);
  float* z2    = allocF(2048L * 512);
  float* pg    = allocF(2048L * 64);
  float* sc    = allocF(2048L * 1536);
  float* hbuf  = allocF(8L * 2048 * 256);
  float* kv1   = allocF(16384L * 320);
  float* kv2   = allocF(16384L * 512);

  _Float16* wsh = (_Float16*)(wsf + fo);
  long ho = 0;
  auto allocH = [&](long n) {
    long np = (n + 15) & ~15L;
    HB b = { wsh + ho, wsh + ho + np };
    ho += 2 * np;
    return b;
  };
  // weights (split once per launch)
  M.wkv1 = allocH(320L * 128);   M.wkv2 = allocH(512L * 320);
  M.wq   = allocH(512L * 512);   M.wo   = allocH(512L * 512);  M.wv = allocH(512L * 512);
  S.wkv1 = allocH(320L * 128);   S.wkv2 = allocH(512L * 320);
  S.wq   = allocH(512L * 512);   S.wo   = allocH(512L * 512);  S.wv = allocH(512L * 512);
  HB wpl1 = allocH(320L * 512),  wpl2 = allocH(128L * 320);
  HB wbb1 = allocH(512L * 192),  wbb2 = allocH(512L * 512);
  HB wlog = allocH(64L * 512),   wscl = allocH(1536L * 512);
  HB waw[4] = { allocH(8L * 256 * 192), allocH(8L * 256 * 256),
                allocH(8L * 256 * 256), allocH(8L * 32 * 256) };
  M.qt = allocH(32L * 512);      S.qt = allocH(32L * 512);
  M.wpl1 = S.wpl1 = wpl1;        M.wpl2 = S.wpl2 = wpl2;
  // activations
  HB embc = allocH(16384L * 128), kv1h = allocH(16384L * 320);
  HB wctx = allocH(2048L * 2048), atto = allocH(2048L * 512), oprj = allocH(2048L * 512);
  HB p1h  = allocH(2048L * 320);
  HB ctxh = allocH(2048L * 192);
  HB z1h  = allocH(2048L * 512), z2h = allocH(2048L * 512);
  HB xh[3] = { allocH(2048L * 192), allocH(2048L * 256), allocH(2048L * 256) };

  // ---- pre-split all GEMM weight operands (pad K->%32, per-expert O->%32) ----
  split(IN(22), 320, 128, 1, 320, 128, M.wkv1, stream);
  split(IN(24), 512, 320, 1, 512, 320, M.wkv2, stream);
  split(IN(42), 512, 512, 1, 512, 512, M.wq, stream);
  split(IN(40), 512, 512, 1, 512, 512, M.wo, stream);
  split(IN(44), 512, 512, 1, 512, 512, M.wv, stream);
  split(IN(30), 320, 128, 1, 320, 128, S.wkv1, stream);
  split(IN(32), 512, 320, 1, 512, 320, S.wkv2, stream);
  split(IN(50), 512, 512, 1, 512, 512, S.wq, stream);
  split(IN(48), 512, 512, 1, 512, 512, S.wo, stream);
  split(IN(52), 512, 512, 1, 512, 512, S.wv, stream);
  split(IN(54), 320, 512, 1, 320, 512, wpl1, stream);
  split(IN(56), 128, 320, 1, 128, 320, wpl2, stream);
  split(bb1W, 512, 176, 1, 512, 192, wbb1, stream);
  split(bb2W, 512, 512, 1, 512, 512, wbb2, stream);
  split(logW, 64, 512, 1, 64, 512, wlog, stream);
  split(scW, 1536, 512, 1, 1536, 512, wscl, stream);
  split(aW[0], 256, 176, 8, 256, 192, waw[0], stream);
  split(aW[1], 256, 256, 8, 256, 256, waw[1], stream);
  split(aW[2], 256, 256, 8, 256, 256, waw[2], stream);
  split(aW[3], 12, 256, 8, 32, 256, waw[3], stream);
  split(IN(60), 1, 512, 1, 32, 512, M.qt, stream);
  split(IN(61), 1, 512, 1, 32, 512, S.qt, stream);
#undef IN

  FuseWs FW = { s_att, qh, qk, p1, kv1, kv2, embc, kv1h, wctx, atto, oprj, p1h };
  float* out = (float*)d_out;

  // ---- fuse(mod) -> emb_mod -> gate (pg, sc) ----
  run_fuse(obs_embed, M, FW, emb, stream);
  concat_split_kernel<<<(2048 * 192 + 255) / 256, 256, 0, stream>>>(obs_state, emb,
                                                                    ctxh.hi, ctxh.lo);
  gemm(ctxh, 192, 0, wbb1, 0, bb1b, 0, z1, 512, 0, HBNULL, 0, 0, 2048, 512, 192, 1, stream);
  ln_act_kernel<<<2048, 128, 0, stream>>>(z1, bbln1g, bbln1b, 512, 2048, 1, 0, z1h.hi, z1h.lo);
  gemm(z1h, 512, 0, wbb2, 0, bb2b, 0, z2, 512, 0, HBNULL, 0, 0, 2048, 512, 512, 1, stream);
  ln_act_kernel<<<2048, 128, 0, stream>>>(z2, bbln2g, bbln2b, 512, 2048, 1, 0, z2h.hi, z2h.lo);
  gemm(z2h, 512, 0, wlog, 0, logb, 0, pg, 64, 0, HBNULL, 0, 0, 2048, 64, 512, 1, stream);
  softmax8_kernel<<<(2048 * 8 + 255) / 256, 256, 0, stream>>>(pg);
  gemm(z2h, 512, 0, wscl, 0, scb, 0, sc, 1536, 0, HBNULL, 0, 0, 2048, 1536, 512, 1, stream);

  // ---- fuse(sub) -> emb_sub -> x0 ----
  run_fuse(obs_embed, S, FW, emb, stream);
  concat_split_kernel<<<(2048 * 192 + 255) / 256, 256, 0, stream>>>(obs_state, emb,
                                                                    xh[0].hi, xh[0].lo);

  // ---- 4 gated MoE layers (batched-by-expert WMMA GEMMs) ----
  const int Kd[4]  = {192, 256, 256, 256};   // padded input widths
  const int Od[4]  = {256, 256, 256, 12};
  const int Op[4]  = {256, 256, 256, 32};    // padded expert output rows in W
  HB xin[4] = { xh[0], xh[1], xh[2], xh[1] };
  HB xout3[3] = { xh[1], xh[2], xh[1] };
  for (int l = 0; l < 4; ++l) {
    gemm(xin[l], Kd[l], 0, waw[l], (long)Op[l] * Kd[l], nullptr, 0,
         hbuf, Od[l], 2048L * Od[l], HBNULL, 0, 0, 2048, Od[l], Kd[l], 8, stream);
    if (l < 3)
      moe_combine_kernel<<<(unsigned)((2048L * Od[l] + 255) / 256), 256, 0, stream>>>(
          hbuf, pg, ab[l], sc, xout3[l].hi, xout3[l].lo, nullptr, Od[l], l);
    else
      moe_combine_kernel<<<(unsigned)((2048L * Od[l] + 255) / 256), 256, 0, stream>>>(
          hbuf, pg, ab[l], nullptr, nullptr, nullptr, out, Od[l], l);
  }
}
